// InvariantPointAttention_27582279975358
// MI455X (gfx1250) — matrix operand order repaired and checked by reading, not compiled, not run on hardware
//
#include <hip/hip_runtime.h>
#include <math.h>

#define NRES 768
#define C1   384
#define C2   128
#define Hh   12
#define SQK  16
#define PQK  4
#define SV   16
#define PV   8
#define KQV_COLS 576
#define PT_COLS  576
#define OCAT     2112   // H*C2 + H*SV + PV*H*3 + PV*H
#define NT       (NRES/32)   // 24 j-tiles of 32

typedef __attribute__((ext_vector_type(16))) _Float16 v16h;
typedef __attribute__((ext_vector_type(8)))  float    v8f;

// ---------------------------------------------------------------------------
// Pose prep: normalize quaternion -> R (row major [n][3][3]), trans [n][3]
// ---------------------------------------------------------------------------
__global__ __launch_bounds__(256)
void ipa_prep_pose(const float* __restrict__ recT, float* __restrict__ R,
                   float* __restrict__ trans) {
  int n = blockIdx.x * blockDim.x + threadIdx.x;
  if (n >= NRES) return;
  float w = recT[n*7+0], x = recT[n*7+1], y = recT[n*7+2], z = recT[n*7+3];
  float inv = rsqrtf(w*w + x*x + y*y + z*z);
  w *= inv; x *= inv; y *= inv; z *= inv;
  float* r = R + n*9;
  r[0] = 1.f - 2.f*(y*y + z*z); r[1] = 2.f*(x*y - w*z);       r[2] = 2.f*(x*z + w*y);
  r[3] = 2.f*(x*y + w*z);       r[4] = 1.f - 2.f*(x*x + z*z); r[5] = 2.f*(y*z - w*x);
  r[6] = 2.f*(x*z - w*y);       r[7] = 2.f*(y*z + w*x);       r[8] = 1.f - 2.f*(x*x + y*y);
  trans[n*3+0] = recT[n*7+4];
  trans[n*3+1] = recT[n*7+5];
  trans[n*3+2] = recT[n*7+6];
}

// ---------------------------------------------------------------------------
// Generic WMMA GEMM (NT): C[M,N] = A[M,K] * B[N,K]^T (+bias[n])
// ---------------------------------------------------------------------------
__global__ __launch_bounds__(32)
void ipa_gemm_nt(const float* __restrict__ A, const float* __restrict__ B,
                 const float* __restrict__ bias, float* __restrict__ C,
                 int M, int Ncols, int K) {
  const int m0   = blockIdx.x * 16;
  const int n0   = blockIdx.y * 16;
  const int lane = threadIdx.x;
  const int hf   = lane >> 4;
  const int l16  = lane & 15;
  const float* __restrict__ arow = A + (long)(m0 + l16) * K;
  const float* __restrict__ brow = B + (long)(n0 + l16) * K;
  v8f acc = {};
  for (int k0 = 0; k0 < K; k0 += 32) {
    v16h a, b;
#pragma unroll
    for (int e = 0; e < 8; ++e) {
      a[e]     = (_Float16)arow[k0 + hf*8 + e];
      a[8 + e] = (_Float16)arow[k0 + 16 + hf*8 + e];
    }
#pragma unroll
    for (int e = 0; e < 16; ++e)
      b[e] = (_Float16)brow[k0 + hf*8 + (e & 7) + (e >> 3)*16];
    acc = __builtin_amdgcn_wmma_f32_16x16x32_f16(false, a, false, b,
                                                 (short)0, acc, false, false);
  }
  const int nc = n0 + l16;
  const float bv = bias ? bias[nc] : 0.0f;
#pragma unroll
  for (int r = 0; r < 8; ++r)
    C[(long)(m0 + hf*8 + r) * Ncols + nc] = acc[r] + bv;
}

// ---------------------------------------------------------------------------
// Rotate/translate pt_local, split into k_pt/q_pt/v_pt
// ---------------------------------------------------------------------------
__global__ __launch_bounds__(192)
void ipa_pt_transform(const float* __restrict__ ptl, const float* __restrict__ R,
                      const float* __restrict__ trans, float* __restrict__ qpt,
                      float* __restrict__ kpt, float* __restrict__ vpt) {
  const int n = blockIdx.x;
  const int col = threadIdx.x;
  const int h = col >> 4, t = col & 15;
  float v[3];
#pragma unroll
  for (int d = 0; d < 3; ++d) {
    float s = trans[n*3 + d];
#pragma unroll
    for (int e = 0; e < 3; ++e)
      s += R[n*9 + d*3 + e] * ptl[n*PT_COLS + e*192 + col];
    v[d] = s;
  }
  if (t < PQK) {
#pragma unroll
    for (int d = 0; d < 3; ++d) kpt[((n*3 + d)*Hh + h)*PQK + t] = v[d];
  } else if (t < 2*PQK) {
#pragma unroll
    for (int d = 0; d < 3; ++d) qpt[((n*3 + d)*Hh + h)*PQK + (t - PQK)] = v[d];
  } else {
#pragma unroll
    for (int d = 0; d < 3; ++d) vpt[((n*3 + d)*Hh + h)*PV + (t - 2*PQK)] = v[d];
  }
}

__global__ __launch_bounds__(256)
void ipa_qn_kn(const float* __restrict__ qpt, const float* __restrict__ kpt,
               float* __restrict__ qn, float* __restrict__ kn) {
  int idx = blockIdx.x * blockDim.x + threadIdx.x;
  if (idx >= NRES * Hh) return;
  int h = idx % Hh, n = idx / Hh;
  float sq = 0.f, sk = 0.f;
#pragma unroll
  for (int d = 0; d < 3; ++d)
#pragma unroll
    for (int p = 0; p < PQK; ++p) {
      float q = qpt[((n*3 + d)*Hh + h)*PQK + p];
      float k = kpt[((n*3 + d)*Hh + h)*PQK + p];
      sq += q*q; sk += k*k;
    }
  qn[idx] = sq; kn[idx] = sk;
}

// ---------------------------------------------------------------------------
// Fused flash-style attention: per query row i, single pass over rep_2d.
//  Stage A (waves 0-1): WMMA rr tile + epilogue -> raw logits s (LDS + wbuf)
//  Stage B (wave 0):    online softmax update (m, S, tile weights, rescale)
//  Stage C (all waves): rescale o2d accumulators, WMMA-accumulate w~^T*rep2d
// Writes: wbuf raw logits, Mbuf/Sbuf, normalized o2d into out_cat.
// ---------------------------------------------------------------------------
__global__ __launch_bounds__(256)
void ipa_attn_fused(const float* __restrict__ rep2d, const float* __restrict__ Wrr,
                    const float* __restrict__ kqv, const float* __restrict__ qpt,
                    const float* __restrict__ kpt, const float* __restrict__ qn,
                    const float* __restrict__ kn, float* __restrict__ wbuf,
                    float* __restrict__ Mbuf, float* __restrict__ Sbuf,
                    float* __restrict__ ocat) {
  const int i   = blockIdx.x;
  const int tid = threadIdx.x;
  const int wv  = tid >> 5;
  const int lane = tid & 31;
  const int hf = lane >> 4, l16 = lane & 15;
  const float Wc = 0.23570226039551584f;   // sqrt(2/(9*PQK))
  const float Wl = 0.57735026918962576f;   // sqrt(1/3)

  __shared__ float s_tile[32][16];   // raw logits for current j-tile
  __shared__ float w_tile[32][16];   // exp weights for current j-tile
  __shared__ float sc_s[16];         // per-head rescale for this tile
  __shared__ float m_s[16], S_s[16]; // running max / running sum
  __shared__ float q1s[Hh][SQK];
  __shared__ float qps[Hh][3*PQK];
  __shared__ float qni_s[Hh];

  // per-i query data into LDS
  if (tid < Hh*SQK)
    q1s[tid >> 4][tid & 15] = kqv[i*KQV_COLS + (tid >> 4)*48 + SQK + (tid & 15)];
  if (tid < Hh*3*PQK) {
    int h = tid / (3*PQK), dp = tid % (3*PQK), d = dp / PQK, p = dp % PQK;
    qps[h][dp] = qpt[((i*3 + d)*Hh + h)*PQK + p];
  }
  if (tid < Hh) qni_s[tid] = qn[i*Hh + tid];
  if (tid < 16) { m_s[tid] = -1e30f; S_s[tid] = 0.f; }
  __syncthreads();

  v8f oacc = {};                       // o2d accumulator (rows=h, cols=c tile)
  const int n0 = wv * 16;              // this wave's c-tile

  for (int jt = 0; jt < NT; ++jt) {
    // ---- Stage A: logits for 32 j's (waves 0,1: 16 each) -----------------
    if (wv < 2) {
      const int j0w = jt*32 + wv*16;
      const float* __restrict__ arow = rep2d + ((long)i*NRES + j0w + l16)*C2;
      const float* __restrict__ brow = Wrr + (l16 < Hh ? l16 : 0)*C2;
      const float bmask = (l16 < Hh) ? 1.0f : 0.0f;
      v8f racc = {};
      for (int k0 = 0; k0 < C2; k0 += 32) {
        v16h a, b;
#pragma unroll
        for (int e = 0; e < 8; ++e) {
          a[e]     = (_Float16)arow[k0 + hf*8 + e];
          a[8 + e] = (_Float16)arow[k0 + 16 + hf*8 + e];
        }
#pragma unroll
        for (int e = 0; e < 16; ++e)
          b[e] = (_Float16)(bmask * brow[k0 + hf*8 + (e & 7) + (e >> 3)*16]);
        racc = __builtin_amdgcn_wmma_f32_16x16x32_f16(false, a, false, b,
                                                      (short)0, racc, false, false);
      }
      if (jt + 1 < NT)   // pull next tile's rows toward L2 while we compute
        __builtin_prefetch(rep2d + ((long)i*NRES + jt*32 + 32 + wv*16 + l16)*C2, 0, 1);
      const int h = l16;
      if (h < Hh) {
        const float qni_v = qni_s[h];
#pragma unroll
        for (int r = 0; r < 8; ++r) {
          const int jl = wv*16 + hf*8 + r;
          const int j  = jt*32 + jl;
          float qk = 0.f;
#pragma unroll
          for (int c = 0; c < SQK; ++c) qk += q1s[h][c] * kqv[j*KQV_COLS + h*48 + c];
          float cross = 0.f;
#pragma unroll
          for (int d = 0; d < 3; ++d)
#pragma unroll
            for (int p = 0; p < PQK; ++p)
              cross += qps[h][d*PQK + p] * kpt[((j*3 + d)*Hh + h)*PQK + p];
          const float d2 = qni_v + kn[j*Hh + h] - 2.0f*cross;
          const float s = Wl * (-Wc*d2 + racc[r] + qk*0.25f);
          s_tile[jl][h] = s;
          wbuf[((long)i*Hh + h)*NRES + j] = s;   // raw logits for o1d/og path
        }
      }
    }
    __syncthreads();

    // ---- Stage B: online softmax update (wave 0, 16 lanes) ---------------
    if (wv == 0 && lane < 16) {
      const int h = lane;
      if (h < Hh) {
        float mold = m_s[h];
        float tmax = mold;
#pragma unroll
        for (int jl = 0; jl < 32; ++jl) tmax = fmaxf(tmax, s_tile[jl][h]);
        const float scale = __expf(mold - tmax);
        float ssum = 0.f;
#pragma unroll
        for (int jl = 0; jl < 32; ++jl) {
          float w = __expf(s_tile[jl][h] - tmax);
          w_tile[jl][h] = w;
          ssum += w;
        }
        S_s[h] = S_s[h]*scale + ssum;
        m_s[h] = tmax;
        sc_s[h] = scale;
      } else {
        sc_s[h] = 1.0f;
#pragma unroll
        for (int jl = 0; jl < 32; ++jl) w_tile[jl][h] = 0.f;
      }
    }
    __syncthreads();

    // ---- Stage C: o2d accumulate (all 8 waves, one c-tile each) ----------
    {
#pragma unroll
      for (int r = 0; r < 8; ++r) oacc[r] *= sc_s[hf*8 + r];
      v16h a, b;
#pragma unroll
      for (int e = 0; e < 8; ++e) {
        a[e]     = (_Float16)w_tile[hf*8 + e][l16];        // A[h][j0..7 /8..15]
        a[8 + e] = (_Float16)w_tile[16 + hf*8 + e][l16];   // A[h][j16..23/24..31]
      }
#pragma unroll
      for (int e = 0; e < 16; ++e)                          // rows just read: L2-hot
        b[e] = (_Float16)rep2d[((long)i*NRES + jt*32 + hf*8 + (e & 7) + (e >> 3)*16)*C2 + n0 + l16];
      oacc = __builtin_amdgcn_wmma_f32_16x16x32_f16(false, a, false, b,
                                                    (short)0, oacc, false, false);
    }
    __syncthreads();
  }

  // ---- epilogue: normalize by running sum, store o2d + (m,S) -------------
#pragma unroll
  for (int r = 0; r < 8; ++r) {
    const int h = hf*8 + r;
    if (h < Hh) ocat[(long)i*OCAT + h*C2 + n0 + l16] = oacc[r] / S_s[h];
  }
  if (tid < Hh) { Mbuf[i*Hh + tid] = m_s[tid]; Sbuf[i*Hh + tid] = S_s[tid]; }
}

// Normalize raw logits into softmax weights: w = exp(s - m) / S
__global__ __launch_bounds__(256)
void ipa_norm_w(float* __restrict__ wbuf, const float* __restrict__ Mbuf,
                const float* __restrict__ Sbuf) {
  const long row = blockIdx.x;   // i*Hh + h
  float* r = wbuf + row * NRES;
  const float m = Mbuf[row];
  const float inv = 1.0f / Sbuf[row];
  for (int j = threadIdx.x; j < NRES; j += 256)
    r[j] = __expf(r[j] - m) * inv;
}

// o1d[i,h,c] = sum_j w * v1d[j,h,c]
__global__ __launch_bounds__(256)
void ipa_o1d(const float* __restrict__ wgt, const float* __restrict__ kqv,
             float* __restrict__ ocat) {
  int idx = blockIdx.x * blockDim.x + threadIdx.x;
  if (idx >= NRES * Hh * SV) return;
  const int c = idx & 15, h = (idx >> 4) % Hh, i = idx / (Hh * SV);
  const float* __restrict__ w = wgt + ((long)i*Hh + h) * NRES;
  float s = 0.f;
  for (int j = 0; j < NRES; ++j) s += w[j] * kqv[j*KQV_COLS + h*48 + 32 + c];
  ocat[(long)i*OCAT + Hh*C2 + h*SV + c] = s;
}

// og[i,d,h,p] = sum_j w * v_pt[j,d,h,p]
__global__ __launch_bounds__(256)
void ipa_og(const float* __restrict__ wgt, const float* __restrict__ vpt,
            float* __restrict__ og) {
  int idx = blockIdx.x * blockDim.x + threadIdx.x;
  if (idx >= NRES * 3 * Hh * PV) return;
  const int p = idx & 7;
  int t = idx >> 3;
  const int h = t % Hh; t /= Hh;
  const int d = t % 3;  const int i = t / 3;
  const float* __restrict__ w = wgt + ((long)i*Hh + h) * NRES;
  float s = 0.f;
  for (int j = 0; j < NRES; ++j) s += w[j] * vpt[((j*3 + d)*Hh + h)*PV + p];
  og[((i*3 + d)*Hh + h)*PV + p] = s;
}

// invert frames -> o_pt [p,h,e] and o_norm [p,h] into out_cat
__global__ __launch_bounds__(256)
void ipa_finalize_pt(const float* __restrict__ og, const float* __restrict__ R,
                     const float* __restrict__ trans, float* __restrict__ ocat) {
  int idx = blockIdx.x * blockDim.x + threadIdx.x;
  if (idx >= NRES * PV * Hh) return;
  const int h = idx % Hh, p = (idx / Hh) % PV, i = idx / (PV * Hh);
  float diff[3];
#pragma unroll
  for (int d = 0; d < 3; ++d)
    diff[d] = og[((i*3 + d)*Hh + h)*PV + p] - trans[i*3 + d];
  float nrm = 0.f;
#pragma unroll
  for (int e = 0; e < 3; ++e) {
    float ol = R[i*9 + 0*3 + e]*diff[0] + R[i*9 + 1*3 + e]*diff[1]
             + R[i*9 + 2*3 + e]*diff[2];
    ocat[(long)i*OCAT + Hh*C2 + Hh*SV + (p*Hh + h)*3 + e] = ol;
    nrm += ol * ol;
  }
  ocat[(long)i*OCAT + Hh*C2 + Hh*SV + PV*Hh*3 + p*Hh + h] = sqrtf(nrm);
}

// ---------------------------------------------------------------------------
extern "C" void kernel_launch(void* const* d_in, const int* in_sizes, int n_in,
                              void* d_out, int out_size, void* d_ws, size_t ws_size,
                              hipStream_t stream) {
  (void)in_sizes; (void)n_in; (void)out_size; (void)ws_size;
  const float* rec_1d  = (const float*)d_in[0];
  const float* rep_2d  = (const float*)d_in[1];
  const float* rec_T   = (const float*)d_in[2];
  const float* W_kqv1d = (const float*)d_in[3];
  const float* W_kqvpt = (const float*)d_in[4];
  const float* W_rr    = (const float*)d_in[5];
  const float* W_final = (const float*)d_in[6];
  const float* b_final = (const float*)d_in[7];
  float* out = (float*)d_out;

  float* ws    = (float*)d_ws;
  float* R     = ws;
  float* trans = R     + NRES*9;
  float* kqv   = trans + NRES*3;
  float* ptl   = kqv   + NRES*KQV_COLS;
  float* qpt   = ptl   + NRES*PT_COLS;
  float* kpt   = qpt   + NRES*3*Hh*PQK;
  float* vpt   = kpt   + NRES*3*Hh*PQK;
  float* qn    = vpt   + NRES*3*Hh*PV;
  float* kn    = qn    + NRES*Hh;
  float* wbuf  = kn    + NRES*Hh;              // raw logits -> weights, 768*12*768
  float* Mbuf  = wbuf  + (long)NRES*Hh*NRES;
  float* Sbuf  = Mbuf  + NRES*Hh;
  float* og    = Sbuf  + NRES*Hh;
  float* ocat  = og    + NRES*3*Hh*PV;

  ipa_prep_pose<<<dim3((NRES + 255)/256), dim3(256), 0, stream>>>(rec_T, R, trans);
  ipa_gemm_nt<<<dim3(NRES/16, KQV_COLS/16), dim3(32), 0, stream>>>(
      rec_1d, W_kqv1d, nullptr, kqv, NRES, KQV_COLS, C1);
  ipa_gemm_nt<<<dim3(NRES/16, PT_COLS/16), dim3(32), 0, stream>>>(
      rec_1d, W_kqvpt, nullptr, ptl, NRES, PT_COLS, C1);
  ipa_pt_transform<<<dim3(NRES), dim3(192), 0, stream>>>(ptl, R, trans, qpt, kpt, vpt);
  ipa_qn_kn<<<dim3((NRES*Hh + 255)/256), dim3(256), 0, stream>>>(qpt, kpt, qn, kn);
  // single pass over rep_2d: logits + online softmax + o2d
  ipa_attn_fused<<<dim3(NRES), dim3(256), 0, stream>>>(
      rep_2d, W_rr, kqv, qpt, kpt, qn, kn, wbuf, Mbuf, Sbuf, ocat);
  ipa_norm_w<<<dim3(NRES*Hh), dim3(256), 0, stream>>>(wbuf, Mbuf, Sbuf);
  ipa_o1d<<<dim3((NRES*Hh*SV + 255)/256), dim3(256), 0, stream>>>(wbuf, kqv, ocat);
  ipa_og<<<dim3((NRES*3*Hh*PV + 255)/256), dim3(256), 0, stream>>>(wbuf, vpt, og);
  ipa_finalize_pt<<<dim3((NRES*PV*Hh + 255)/256), dim3(256), 0, stream>>>(og, R, trans, ocat);
  ipa_gemm_nt<<<dim3(NRES/16, C1/16), dim3(32), 0, stream>>>(
      ocat, W_final, b_final, out, NRES, C1, OCAT);
}
